// RGCN_18880676233382
// MI455X (gfx1250) — compile-verified, matrix operand order checked by reading, not run on hardware
//
#include <hip/hip_runtime.h>

typedef __attribute__((ext_vector_type(16))) _Float16 v16h;
typedef __attribute__((ext_vector_type(8)))  float    v8f;

#define R_REL 16
#define DIN   386
#define HID   256
#define K1TOT (R_REL * DIN + DIN)   // 6562 = 16 relation blocks + root block
#define AK    6592                  // K1TOT padded to a multiple of BK
#define BM    128
#define BN    128
#define BK    64
#define KT    (AK / BK)             // 103

// ---------------------------------------------------------------- utilities
__global__ __launch_bounds__(256) void k_zero_f32(float* __restrict__ p, size_t n) {
  size_t i = (size_t)blockIdx.x * blockDim.x + threadIdx.x;
  size_t stride = (size_t)gridDim.x * blockDim.x;
  for (; i < n; i += stride) p[i] = 0.0f;
}

__global__ __launch_bounds__(256) void k_count(const int* __restrict__ dst,
                                               const int* __restrict__ et,
                                               float* __restrict__ cnt, int E, int N) {
  int e = blockIdx.x * blockDim.x + threadIdx.x;
  if (e < E) atomicAdd(&cnt[(size_t)et[e] * N + dst[e]], 1.0f);
}

// one block per edge: scatter x[src] into sums[(etype*N+dst)*DIN + d]
__global__ __launch_bounds__(256) void k_scatter1(const float* __restrict__ x,
                                                  const int* __restrict__ src,
                                                  const int* __restrict__ dst,
                                                  const int* __restrict__ et,
                                                  float* __restrict__ sums, int N) {
  int e = blockIdx.x;
  int s = src[e], d = dst[e], r = et[e];
  const float* xs = x + (size_t)s * DIN;
  float* sp = sums + ((size_t)r * N + d) * DIN;
  int t = threadIdx.x;
  atomicAdd(&sp[t], xs[t]);
  if (t < DIN - 256) atomicAdd(&sp[256 + t], xs[256 + t]);
}

// Build padded f16 A matrix: A[n, k] = mean(r, n, d) for k = r*DIN+d, x[n, d] for root block
__global__ __launch_bounds__(256) void k_build_A(const float* __restrict__ sums,
                                                 const float* __restrict__ cnt,
                                                 const float* __restrict__ x,
                                                 _Float16* __restrict__ Ah, int N) {
  int k = blockIdx.x * 256 + threadIdx.x;
  int n = blockIdx.y;
  if (k >= AK) return;
  float v = 0.0f;
  if (n < N && k < K1TOT) {
    if (k < R_REL * DIN) {
      int r = k / DIN, d = k - r * DIN;
      float c = cnt[(size_t)r * N + n];
      if (c < 1.0f) c = 1.0f;
      v = sums[((size_t)r * N + n) * DIN + d] / c;
    } else {
      v = x[(size_t)n * DIN + (k - R_REL * DIN)];
    }
  }
  Ah[(size_t)n * AK + k] = (_Float16)v;
}

// Build pre-TRANSPOSED padded f16 B: Bt[col][k] so GEMM stages it with contiguous async b128
__global__ __launch_bounds__(256) void k_build_Bt(const float* __restrict__ W1,
                                                  const float* __restrict__ root1,
                                                  _Float16* __restrict__ Bt) {
  size_t idx = (size_t)blockIdx.x * 256 + threadIdx.x;
  if (idx >= (size_t)HID * AK) return;
  int col = (int)(idx / AK), k = (int)(idx % AK);
  float v = 0.0f;
  if (k < R_REL * DIN)      v = W1[(size_t)k * HID + col];
  else if (k < K1TOT)       v = root1[(size_t)(k - R_REL * DIN) * HID + col];
  Bt[idx] = (_Float16)v;
}

// ---------------------------------------------------------------- WMMA GEMM
__device__ __forceinline__ void async_b128(unsigned lds, unsigned long long ga) {
  asm volatile("global_load_async_to_lds_b128 %0, %1, off" :: "v"(lds), "v"(ga) : "memory");
}

__global__ __launch_bounds__(256) void k_gemm1(const _Float16* __restrict__ Ah,
                                               const _Float16* __restrict__ Bt,
                                               const float* __restrict__ b1,
                                               float* __restrict__ h, int Nrows) {
  __shared__ __align__(16) _Float16 sA[2][BM][BK];    // 16KB per buffer
  __shared__ __align__(16) _Float16 sBt[2][BN][BK];   // 16KB per buffer
  const int tid = threadIdx.x;
  const int lane = tid & 31;
  const int wid = tid >> 5;
  const int wm = wid & 3;   // 4 wave-rows of 32
  const int wn = wid >> 2;  // 2 wave-cols of 64
  const int row0 = blockIdx.x * BM;
  const int col0 = blockIdx.y * BN;

  v8f acc[2][4] = {};

  const int lrow = lane & 15;
  const int kb = (lane < 16) ? 0 : 8;   // ISA 16-bit A/B lane->K mapping

  // ---- per-thread staging bases: each thread owns 4 A chunks + 4 B chunks (16B each)
  const int q   = tid >> 3;   // 0..31  (row of A tile / col of Bt tile, +32*p)
  const int seg = tid & 7;    // 16-byte segment within a 64-half line
  const _Float16* gA = Ah + (size_t)(row0 + q) * AK + seg * 8;
  const _Float16* gB = Bt + (size_t)(col0 + q) * AK + seg * 8;
  const unsigned ldsA0 = (unsigned)(size_t)&sA[0][q][seg * 8];
  const unsigned ldsB0 = (unsigned)(size_t)&sBt[0][q][seg * 8];
  const unsigned bufA = (unsigned)((size_t)&sA[1][0][0] - (size_t)&sA[0][0][0]);
  const unsigned bufB = (unsigned)((size_t)&sBt[1][0][0] - (size_t)&sBt[0][0][0]);
  const unsigned long long rstride = (unsigned long long)32 * AK * sizeof(_Float16);

  auto stage = [&](int buf, int k0) {
    unsigned la = ldsA0 + (unsigned)buf * bufA;
    unsigned lb = ldsB0 + (unsigned)buf * bufB;
    unsigned long long ga = (unsigned long long)(size_t)(gA + k0);
    unsigned long long gb = (unsigned long long)(size_t)(gB + k0);
    __builtin_prefetch(gA + k0 + BK, 0, 1);   // global_prefetch_b8: tile after next
#pragma unroll
    for (int p = 0; p < 4; ++p) {             // rows/cols q, q+32, q+64, q+96
      async_b128(la + p * 4096u, ga + (unsigned long long)p * rstride);
      async_b128(lb + p * 4096u, gb + (unsigned long long)p * rstride);
    }
  };

  stage(0, 0);
  asm volatile("s_wait_asynccnt 0" ::: "memory");
  __syncthreads();

  for (int kt = 0; kt < KT; ++kt) {
    const int cur = kt & 1;
    if (kt + 1 < KT) stage(cur ^ 1, (kt + 1) * BK);

#pragma unroll
    for (int kk = 0; kk < BK; kk += 32) {
      v16h afrag[2], bfrag[4];
#pragma unroll
      for (int f = 0; f < 2; ++f) {
        const _Float16* ap = &sA[cur][wm * 32 + f * 16 + lrow][kk + kb];
        union { uint4 u[2]; v16h v; } t;
        t.u[0] = *(const uint4*)ap;          // K = kk+kb .. +7
        t.u[1] = *(const uint4*)(ap + 16);   // K = kk+kb+16 .. +23
        afrag[f] = t.v;
      }
#pragma unroll
      for (int f = 0; f < 4; ++f) {
        const _Float16* bp = &sBt[cur][wn * 64 + f * 16 + lrow][kk + kb];
        union { uint4 u[2]; v16h v; } t;
        t.u[0] = *(const uint4*)bp;
        t.u[1] = *(const uint4*)(bp + 16);
        bfrag[f] = t.v;
      }
#pragma unroll
      for (int a = 0; a < 2; ++a)
#pragma unroll
        for (int b = 0; b < 4; ++b)
          acc[a][b] = __builtin_amdgcn_wmma_f32_16x16x32_f16(
              false, afrag[a], false, bfrag[b], (short)0, acc[a][b], false, false);
    }

    asm volatile("s_wait_asynccnt 0" ::: "memory");
    __syncthreads();
  }

  // epilogue: C layout: VGPR j -> M=j (lanes 0-15) / M=8+j (lanes 16-31), N=lane%16
  const int moff = (lane < 16) ? 0 : 8;
#pragma unroll
  for (int a = 0; a < 2; ++a)
#pragma unroll
    for (int b = 0; b < 4; ++b) {
      int col = col0 + wn * 64 + b * 16 + (lane & 15);
      float bias = b1[col];
#pragma unroll
      for (int j = 0; j < 8; ++j) {
        int row = row0 + wm * 32 + a * 16 + j + moff;
        if (row < Nrows) {
          float v = acc[a][b][j] + bias;
          h[(size_t)row * HID + col] = v > 0.0f ? v : 0.0f;  // ReLU
        }
      }
    }
}

// ---------------------------------------------------------------- layer 2
__global__ __launch_bounds__(256) void k_scatter2(const float* __restrict__ h,
                                                  const int* __restrict__ src,
                                                  const int* __restrict__ dst,
                                                  const int* __restrict__ et,
                                                  float* __restrict__ sums2, int N) {
  int e = blockIdx.x;
  int s = src[e], d = dst[e], r = et[e];
  int t = threadIdx.x;
  atomicAdd(&sums2[((size_t)r * N + d) * HID + t], h[(size_t)s * HID + t]);
}

__global__ __launch_bounds__(256) void k_layer2(const float* __restrict__ sums2,
                                                const float* __restrict__ cnt,
                                                const float* __restrict__ h,
                                                const float* __restrict__ W2,
                                                const float* __restrict__ root2,
                                                const float* __restrict__ b2,
                                                float* __restrict__ out, int N) {
  int n = blockIdx.x, t = threadIdx.x;
  float a0 = 0.0f, a1 = 0.0f;
#pragma unroll
  for (int r = 0; r < R_REL; ++r) {
    float c = cnt[(size_t)r * N + n];
    if (c < 1.0f) c = 1.0f;
    float m = sums2[((size_t)r * N + n) * HID + t] / c;
    a0 += m * W2[r * HID * 2 + t * 2 + 0];
    a1 += m * W2[r * HID * 2 + t * 2 + 1];
  }
  float hv = h[(size_t)n * HID + t];
  a0 += hv * root2[t * 2 + 0];
  a1 += hv * root2[t * 2 + 1];
  __shared__ float red[2][256];
  red[0][t] = a0; red[1][t] = a1;
  __syncthreads();
  for (int s = 128; s > 0; s >>= 1) {
    if (t < s) { red[0][t] += red[0][t + s]; red[1][t] += red[1][t + s]; }
    __syncthreads();
  }
  if (t == 0) {
    out[(size_t)n * 2 + 0] = tanhf(red[0][0] + b2[0]);
    out[(size_t)n * 2 + 1] = tanhf(red[1][0] + b2[1]);
  }
}

// ---------------------------------------------------------------- launcher
extern "C" void kernel_launch(void* const* d_in, const int* in_sizes, int n_in,
                              void* d_out, int out_size, void* d_ws, size_t ws_size,
                              hipStream_t stream) {
  const float* x     = (const float*)d_in[0];
  const int*   ei    = (const int*)d_in[1];   // [2, E] flattened
  const int*   et    = (const int*)d_in[2];
  const float* W1    = (const float*)d_in[3];
  const float* root1 = (const float*)d_in[4];
  const float* b1    = (const float*)d_in[5];
  const float* W2    = (const float*)d_in[6];
  const float* root2 = (const float*)d_in[7];
  const float* b2    = (const float*)d_in[8];
  float* out = (float*)d_out;

  const int N = in_sizes[0] / DIN;
  const int E = in_sizes[1] / 2;
  const int* src = ei;
  const int* dst = ei + E;
  const int Mpad = ((N + BM - 1) / BM) * BM;

  // workspace carve-out (256B aligned)
  char* w = (char*)d_ws;
  auto carve = [&](size_t bytes) -> char* {
    char* p = w; w += (bytes + 255) & ~(size_t)255; return p;
  };
  float*     cnt  = (float*)carve((size_t)R_REL * N * sizeof(float));
  float*     sums = (float*)carve((size_t)R_REL * N * DIN * sizeof(float)); // reused for layer 2
  _Float16*  Ah   = (_Float16*)carve((size_t)Mpad * AK * sizeof(_Float16));
  _Float16*  Bt   = (_Float16*)carve((size_t)HID * AK * sizeof(_Float16));
  float*     h    = (float*)carve((size_t)N * HID * sizeof(float));

  dim3 blk(256);
  auto zgrid = [](size_t n) {
    size_t g = (n + 255) / 256;
    if (g > (size_t)1 << 20) g = (size_t)1 << 20;
    return dim3((unsigned)g);
  };

  // ---- layer 1 ----
  k_zero_f32<<<zgrid((size_t)R_REL * N), blk, 0, stream>>>(cnt, (size_t)R_REL * N);
  k_zero_f32<<<zgrid((size_t)R_REL * N * DIN), blk, 0, stream>>>(sums, (size_t)R_REL * N * DIN);
  k_count<<<(E + 255) / 256, blk, 0, stream>>>(dst, et, cnt, E, N);
  k_scatter1<<<E, blk, 0, stream>>>(x, src, dst, et, sums, N);
  k_build_Bt<<<(unsigned)(((size_t)HID * AK + 255) / 256), blk, 0, stream>>>(W1, root1, Bt);
  k_build_A<<<dim3((AK + 255) / 256, Mpad), blk, 0, stream>>>(sums, cnt, x, Ah, N);
  k_gemm1<<<dim3(Mpad / BM, HID / BN), blk, 0, stream>>>(Ah, Bt, b1, h, N);

  // ---- layer 2 (reuse sums buffer; counts are identical) ----
  k_zero_f32<<<zgrid((size_t)R_REL * N * HID), blk, 0, stream>>>(sums, (size_t)R_REL * N * HID);
  k_scatter2<<<E, blk, 0, stream>>>(h, src, dst, et, sums, N);
  k_layer2<<<N, blk, 0, stream>>>(sums, cnt, h, W2, root2, b2, out, N);
  (void)n_in; (void)out_size; (void)ws_size;
}
